// BiLSTMCRF_9689446220424
// MI455X (gfx1250) — matrix operand order repaired and checked by reading, not compiled, not run on hardware
//
#include <hip/hip_runtime.h>
#include <hip/hip_bf16.h>

#define SL    256   // sequence length
#define NB    64    // batch
#define EMBD  256
#define HD    256   // hidden per direction
#define HIDT  512
#define GATES 1024  // 4*HD
#define NTAGS 9

typedef __attribute__((ext_vector_type(16))) __bf16 v16bf;
typedef __attribute__((ext_vector_type(8)))  float  v8f;

union FragBF { v16bf v; uint4 q[2]; };

__device__ __forceinline__ unsigned short f32_to_bf16(float f) {
  unsigned int x = __float_as_uint(f);
  x += 0x7FFFu + ((x >> 16) & 1u);     // round-to-nearest-even
  return (unsigned short)(x >> 16);
}

__device__ __forceinline__ float sigmf(float x) { return 1.0f / (1.0f + __expf(-x)); }
__device__ __forceinline__ float tanh_fast(float x) {
  float e = __expf(2.0f * x);
  return 1.0f - 2.0f / (e + 1.0f);
}

__device__ __forceinline__ v8f wmma_bf16(v16bf a, v16bf b, v8f c) {
  return __builtin_amdgcn_wmma_f32_16x16x32_bf16(false, a, false, b, (short)0, c,
                                                 false, false);
}

// A fragment: 16x32 bf16 tile (M x K). LDS row-major, stride 264 halves.
// lane: m = lane&15, g = lane>>4. u32 i holds K = g*16+2i =>
// q[0] = halves K=g*16..g*16+7, q[1] = halves K=g*16+8..+15  (both 16B contiguous).
__device__ __forceinline__ v16bf load_a_frag(const unsigned short* lds, int mt, int kc, int lane) {
  int row = mt * 16 + (lane & 15);
  int g = lane >> 4;
  const unsigned short* p = lds + row * 264 + kc * 32 + g * 16;
  FragBF f;
  f.q[0] = *(const uint4*)(p);
  f.q[1] = *(const uint4*)(p + 8);
  return f.v;
}

// B fragment: 32x16 bf16 tile (K x N), source stored as W[n][k] row-major in LDS.
// lane: n = lane&15, g = lane>>4. u32 i holds K = g*16+2i => halves K=g*16..g*16+15.
__device__ __forceinline__ v16bf load_b_frag_lds(const unsigned short* lds, int nt, int kc, int lane) {
  int row = nt * 16 + (lane & 15);
  int g = lane >> 4;
  const unsigned short* p = lds + row * 264 + kc * 32 + g * 16;
  FragBF f;
  f.q[0] = *(const uint4*)(p);
  f.q[1] = *(const uint4*)(p + 8);
  return f.v;
}

// B fragment from pre-swizzled global memory: frag = 256 u32, lane reads u32[lane*8..+8]
// => two coalesced b128 loads per lane.
__device__ __forceinline__ v16bf load_b_frag_g(const unsigned int* base, int lane) {
  FragBF f;
  f.q[0] = *(const uint4*)(base + lane * 8);
  f.q[1] = *(const uint4*)(base + lane * 8 + 4);
  return f.v;
}

// ---------------------------------------------------------------------------
// Kernel 1: embedding gather + time-parallel input GEMM for both directions.
// z_x[dir][s][b][0..1023] = emb[inputs[b][s]] @ W_ih[dir]^T + (b_ih+b_hh)[dir]
// M = S*B = 16384, N = 2048 (fwd|bwd), K = 256. WG tile 64x128, 8 waves.
// ---------------------------------------------------------------------------
__global__ __launch_bounds__(256) void k_embed_xgemm(
    const int* __restrict__ inputs, const float* __restrict__ emb,
    const float* __restrict__ w_ih_f, const float* __restrict__ w_ih_b,
    const float* __restrict__ b_ih_f, const float* __restrict__ b_hh_f,
    const float* __restrict__ b_ih_b, const float* __restrict__ b_hh_b,
    float* __restrict__ zx)
{
  __shared__ unsigned short ldsA[64 * 264];
  __shared__ unsigned short ldsB[128 * 264];
  int tid = threadIdx.x;
  int lane = tid & 31, w = tid >> 5;
  int m0 = blockIdx.x * 64;
  int n0 = blockIdx.y * 128;

  // Stage A: gather 64 embedding rows, f32 -> bf16 (coalesced across columns)
  for (int i = tid; i < 64 * 256; i += 256) {
    int r = i >> 8, c = i & 255;
    int m = m0 + r, s = m >> 6, b = m & 63;
    int tok = inputs[b * SL + s];
    ldsA[r * 264 + c] = f32_to_bf16(emb[(size_t)tok * EMBD + c]);
  }
  // Stage B: 128 weight rows (fwd rows 0..1023, bwd rows 1024..2047)
  for (int i = tid; i < 128 * 256; i += 256) {
    int r = i >> 8, c = i & 255;
    int n = n0 + r;
    const float* wr = (n < GATES) ? (w_ih_f + (size_t)n * EMBD)
                                  : (w_ih_b + (size_t)(n - GATES) * EMBD);
    ldsB[r * 264 + c] = f32_to_bf16(wr[c]);
  }
  __syncthreads();

  int mt = w & 3;      // M-tile within WG (16 rows each)
  int ng = w >> 2;     // N-group (4 tiles of 16 each)
  v8f acc[4];
  #pragma unroll
  for (int j = 0; j < 4; ++j)
    #pragma unroll
    for (int e = 0; e < 8; ++e) acc[j][e] = 0.0f;

  #pragma unroll
  for (int kc = 0; kc < 8; ++kc) {
    v16bf af = load_a_frag(ldsA, mt, kc, lane);
    #pragma unroll
    for (int j = 0; j < 4; ++j) {
      v16bf bf = load_b_frag_lds(ldsB, ng * 4 + j, kc, lane);
      acc[j] = wmma_bf16(af, bf, acc[j]);
    }
  }

  // Epilogue: bias add + store f32, layout [dir][m = s*64+b][1024]
  #pragma unroll
  for (int j = 0; j < 4; ++j) {
    int n = n0 + (ng * 4 + j) * 16 + (lane & 15);
    int dir = n >> 10, col = n & 1023;
    const float* bi = dir ? b_ih_b : b_ih_f;
    const float* bh = dir ? b_hh_b : b_hh_f;
    float bias = bi[col] + bh[col];
    int mg0 = m0 + mt * 16 + ((lane >> 4) << 3);
    #pragma unroll
    for (int r = 0; r < 8; ++r) {
      zx[((size_t)dir * (SL * NB) + (size_t)(mg0 + r)) * GATES + col] = acc[j][r] + bias;
    }
  }
}

// ---------------------------------------------------------------------------
// Kernel 1b: convert W_hh (f32, [1024][256]) into bf16 B-fragment-major layout:
// whh_sw[dir][nt(64)][kc(8)] = 256 u32 fragment (lane-major, 8 u32 per lane).
// ---------------------------------------------------------------------------
__global__ __launch_bounds__(256) void k_whh_swz(
    const float* __restrict__ wf, const float* __restrict__ wb,
    unsigned short* __restrict__ out)
{
  int idx = blockIdx.x * 256 + threadIdx.x;   // 0 .. 2*1024*256-1
  int dir = idx >> 18;
  int rem = idx & ((1 << 18) - 1);
  int n = rem >> 8, k = rem & 255;
  float v = (dir ? wb : wf)[(size_t)n * HD + k];
  int nt = n >> 4, nl = n & 15;
  int kc = k >> 5, kk = k & 31;
  int g = kk >> 4, i = (kk & 15) >> 1, h = kk & 1;
  int L = g * 16 + nl;
  size_t pos = ((((size_t)dir * 64 + nt) * 8 + kc) * 256 + L * 8 + i) * 2 + h;
  out[pos] = f32_to_bf16(v);
}

// ---------------------------------------------------------------------------
// Kernel 2: persistent BiLSTM recurrence. One 32-wave block per direction.
// h kept bf16 in LDS (A operand), c in registers, W_hh streamed from L2 in
// swizzled fragment order. Each wave owns 16 batch rows x 32 hidden units,
// i.e. all 4 gates for those units -> gate fusion entirely in registers.
// ---------------------------------------------------------------------------
__global__ __launch_bounds__(1024) void k_bilstm(
    const float* __restrict__ zx, const unsigned int* __restrict__ whh,
    float* __restrict__ hbuf)
{
  __shared__ unsigned short ldsH[64 * 264];
  int tid = threadIdx.x, lane = tid & 31, w = tid >> 5;
  int dir = blockIdx.x;
  int mt = w & 3;        // 16 batch rows
  int ht = w >> 2;       // 32 hidden units (2 N-tiles per gate)

  for (int i = tid; i < 64 * 264; i += 1024) ldsH[i] = 0;
  float c_reg[2][8];
  #pragma unroll
  for (int jt = 0; jt < 2; ++jt)
    #pragma unroll
    for (int r = 0; r < 8; ++r) c_reg[jt][r] = 0.0f;
  __syncthreads();

  int mloc = mt * 16 + ((lane >> 4) << 3);
  int nl = lane & 15;

  for (int t = 0; t < SL; ++t) {
    int s = dir ? (SL - 1 - t) : t;                 // backward walks reversed
    const float* zxs = zx + ((size_t)dir * (SL * NB) + (size_t)s * NB) * GATES;
    if (t + 1 < SL) {
      int s2 = dir ? (SL - 2 - t) : (t + 1);
      __builtin_prefetch(zx + ((size_t)dir * (SL * NB) + (size_t)s2 * NB) * GATES, 0, 1);
    }

    // Init accumulators with precomputed x-part + bias
    v8f acc[4][2];
    #pragma unroll
    for (int g4 = 0; g4 < 4; ++g4)
      #pragma unroll
      for (int jt = 0; jt < 2; ++jt) {
        int n = g4 * 256 + (ht * 2 + jt) * 16 + nl;
        #pragma unroll
        for (int r = 0; r < 8; ++r)
          acc[g4][jt][r] = zxs[(size_t)(mloc + r) * GATES + n];
      }

    // z += h_prev @ W_hh^T : 8 K-chunks x 8 tiles = 64 WMMA per wave per step
    #pragma unroll
    for (int kc = 0; kc < 8; ++kc) {
      v16bf af = load_a_frag(ldsH, mt, kc, lane);
      #pragma unroll
      for (int g4 = 0; g4 < 4; ++g4)
        #pragma unroll
        for (int jt = 0; jt < 2; ++jt) {
          int ntf = g4 * 16 + ht * 2 + jt;
          v16bf bf = load_b_frag_g(whh + (((size_t)dir * 64 + ntf) * 8 + kc) * 256, lane);
          acc[g4][jt] = wmma_bf16(af, bf, acc[g4][jt]);
        }
    }
    __syncthreads();   // all waves done reading ldsH

    // Gate fusion: c = sig(f)*c + sig(i)*tanh(g); h = sig(o)*tanh(c)
    float* hrow = hbuf + ((size_t)dir * SL * NB + (size_t)s * NB) * HD;
    #pragma unroll
    for (int jt = 0; jt < 2; ++jt) {
      int u = (ht * 2 + jt) * 16 + nl;
      #pragma unroll
      for (int r = 0; r < 8; ++r) {
        float zi = acc[0][jt][r], zf = acc[1][jt][r];
        float zg = acc[2][jt][r], zo = acc[3][jt][r];
        float cn = sigmf(zf) * c_reg[jt][r] + sigmf(zi) * tanh_fast(zg);
        c_reg[jt][r] = cn;
        float hv = sigmf(zo) * tanh_fast(cn);
        int m = mloc + r;
        hrow[(size_t)m * HD + u] = hv;
        ldsH[m * 264 + u] = f32_to_bf16(hv);
      }
    }
    __syncthreads();   // new h visible before next step's reads
  }
}

// ---------------------------------------------------------------------------
// Kernel 3: emissions[m][j] = [h_f|h_b] . w_out[j+1] + b_out[j+1], j=0..8
// (reference keeps tags 1..9 only). N is tiny -> plain VALU dot products.
// ---------------------------------------------------------------------------
__global__ __launch_bounds__(256) void k_emissions(
    const float* __restrict__ hbuf, const float* __restrict__ w_out,
    const float* __restrict__ b_out, float* __restrict__ em)
{
  int idx = blockIdx.x * 256 + threadIdx.x;
  if (idx >= SL * NB * NTAGS) return;
  int m = idx / NTAGS, j = idx % NTAGS;
  const float* hf = hbuf + (size_t)m * HD;
  const float* hb = hbuf + (size_t)SL * NB * HD + (size_t)m * HD;
  const float* wo = w_out + (size_t)(j + 1) * HIDT;
  float acc = b_out[j + 1];
  for (int u = 0; u < HD; ++u)
    acc = __builtin_fmaf(hf[u], wo[u], __builtin_fmaf(hb[u], wo[HD + u], acc));
  em[idx] = acc;
}

// ---------------------------------------------------------------------------
// Kernel 4: CRF negative log-likelihood. One thread per batch element,
// 9-state forward algorithm with logsumexp, block reduction to the scalar.
// ---------------------------------------------------------------------------
__global__ __launch_bounds__(64) void k_crf(
    const float* __restrict__ em, const int* __restrict__ tags,
    const unsigned char* __restrict__ mask,
    const float* __restrict__ start_t, const float* __restrict__ end_t,
    const float* __restrict__ trans, float* __restrict__ out)
{
  int b = threadIdx.x;
  float score[NTAGS];
  int t0 = tags[b * SL + 0] - 1;
  float num = start_t[t0] + em[b * NTAGS + t0];
  int prev = t0;
  #pragma unroll
  for (int j = 0; j < NTAGS; ++j) score[j] = start_t[j] + em[b * NTAGS + j];

  for (int s = 1; s < SL; ++s) {
    const float* ems = em + (size_t)(s * NB + b) * NTAGS;
    int msk = mask[b * SL + s];
    int tg = tags[b * SL + s] - 1;
    if (msk) { num += trans[prev * NTAGS + tg] + ems[tg]; prev = tg; }
    float nxt[NTAGS];
    #pragma unroll
    for (int j = 0; j < NTAGS; ++j) {
      float mx = -1e30f;
      #pragma unroll
      for (int i = 0; i < NTAGS; ++i)
        mx = fmaxf(mx, score[i] + trans[i * NTAGS + j]);
      float sum = 0.0f;
      #pragma unroll
      for (int i = 0; i < NTAGS; ++i)
        sum += __expf(score[i] + trans[i * NTAGS + j] - mx);
      nxt[j] = ems[j] + mx + __logf(sum);
    }
    if (msk) {
      #pragma unroll
      for (int j = 0; j < NTAGS; ++j) score[j] = nxt[j];
    }
  }
  num += end_t[prev];
  float mx = -1e30f;
  #pragma unroll
  for (int j = 0; j < NTAGS; ++j) mx = fmaxf(mx, score[j] + end_t[j]);
  float sum = 0.0f;
  #pragma unroll
  for (int j = 0; j < NTAGS; ++j) sum += __expf(score[j] + end_t[j] - mx);
  float den = mx + __logf(sum);

  __shared__ float red[NB];
  red[b] = num - den;
  __syncthreads();
  if (b == 0) {
    float acc = 0.0f;
    for (int i = 0; i < NB; ++i) acc += red[i];
    out[0] = -(acc / (float)NB);
  }
}

// ---------------------------------------------------------------------------
extern "C" void kernel_launch(void* const* d_in, const int* in_sizes, int n_in,
                              void* d_out, int out_size, void* d_ws, size_t ws_size,
                              hipStream_t stream) {
  const int*   inputs  = (const int*)d_in[0];
  const int*   tags    = (const int*)d_in[1];
  const unsigned char* mask = (const unsigned char*)d_in[2];
  const float* emb     = (const float*)d_in[3];
  const float* w_ih_f  = (const float*)d_in[4];
  const float* w_hh_f  = (const float*)d_in[5];
  const float* b_ih_f  = (const float*)d_in[6];
  const float* b_hh_f  = (const float*)d_in[7];
  const float* w_ih_b  = (const float*)d_in[8];
  const float* w_hh_b  = (const float*)d_in[9];
  const float* b_ih_b  = (const float*)d_in[10];
  const float* b_hh_b  = (const float*)d_in[11];
  const float* w_out   = (const float*)d_in[12];
  const float* b_out   = (const float*)d_in[13];
  const float* start_t = (const float*)d_in[14];
  const float* end_t   = (const float*)d_in[15];
  const float* trans   = (const float*)d_in[16];

  // Workspace layout (f32 unless noted):
  float* zx   = (float*)d_ws;                              // [2][S*B][1024]  128 MB
  float* hbuf = zx + (size_t)2 * SL * NB * GATES;          // [2][S*B][256]    32 MB
  float* em   = hbuf + (size_t)2 * SL * NB * HD;           // [S*B][9]        576 KB
  unsigned short* whh = (unsigned short*)(em + (size_t)SL * NB * NTAGS); // bf16, 1 MB

  dim3 g1(SL * NB / 64, 2 * GATES / 128);                  // (256, 16)
  k_embed_xgemm<<<g1, 256, 0, stream>>>(inputs, emb, w_ih_f, w_ih_b,
                                        b_ih_f, b_hh_f, b_ih_b, b_hh_b, zx);
  k_whh_swz<<<(2 * GATES * HD) / 256, 256, 0, stream>>>(w_hh_f, w_hh_b, whh);
  k_bilstm<<<2, 1024, 0, stream>>>(zx, (const unsigned int*)whh, hbuf);
  k_emissions<<<(SL * NB * NTAGS + 255) / 256, 256, 0, stream>>>(hbuf, w_out, b_out, em);
  k_crf<<<1, 64, 0, stream>>>(em, tags, mask, start_t, end_t, trans, (float*)d_out);
}
